// RNNCellWithStateCommunication_73890617360832
// MI455X (gfx1250) — matrix operand order repaired and checked by reading, not compiled, not run on hardware
//
#include <hip/hip_runtime.h>
#include <hip/hip_bf16.h>

// ---------------------------------------------------------------------------
// MI455X (gfx1250) fused RNN-cell-with-state-communication.
// Memory-bound (~77 GFLOP vs ~300MB HBM -> ~13us floor at 23.3 TB/s).
// bf16 activations + v_wmma_f32_16x16x32_bf16 everywhere; weights are
// pre-transposed to [N x K] bf16 so every fragment is a contiguous per-lane
// 16B/32B load matching the CDNA5 WMMA VGPR layouts. B tiles are staged in
// LDS once per block (double-buffered, padded against bank conflicts) instead
// of 8x redundantly per wave; LSTM gate math fused into the last GEMM.
// ---------------------------------------------------------------------------

typedef __attribute__((ext_vector_type(16))) __bf16       v16bf;
typedef __attribute__((ext_vector_type(8)))  float        v8f;
typedef __attribute__((ext_vector_type(4)))  unsigned int v4u;

static constexpr int Bn = 32768;   // batch
static constexpr int Hn = 256;     // hidden
static constexpr int Kd = 512;     // every GEMM in this net has K = 512
static constexpr float kLeaky = 0.2f;

static constexpr int KC   = 128;       // K-chunk staged in LDS
static constexpr int LDSP = 136;       // padded row stride (272B = 68 dwords)

union FragU { v4u u[2]; v16bf v; };

// A fragment: 16x32 bf16, row-major activations [.. x Kd].
// lanes 0-15 -> M=lane, K {k0..k0+7, k0+16..k0+23}
// lanes 16-31 -> M=lane-16, K {k0+8..k0+15, k0+24..k0+31}
__device__ __forceinline__ v16bf load_a_frag(const __bf16* A, int rowBase,
                                             int k0, int lane) {
  const int row  = rowBase + (lane & 15);
  const int koff = k0 + ((lane >> 4) << 3);
  const __bf16* p = A + row * Kd + koff;
  FragU f;
  f.u[0] = *(const v4u*)(p);           // K koff..koff+7   -> VGPR0..3
  f.u[1] = *(const v4u*)(p + 16);      // K koff+16..+23   -> VGPR4..7
  return f.v;
}

// B fragment: 32x16 bf16 from LDS-staged transposed weights.
// lanes 0-15 -> N=lane, K {kk..kk+7, kk+16..kk+23} ; lanes 16-31 -> K {kk+8..kk+15, kk+24..kk+31}
__device__ __forceinline__ v16bf lds_b_frag(const __bf16* buf, int tileRow16,
                                            int kk, int lane) {
  const int r    = tileRow16 + (lane & 15);
  const int koff = kk + ((lane >> 4) << 3);
  const __bf16* p = buf + r * LDSP + koff;
  FragU f;
  f.u[0] = *(const v4u*)(p);
  f.u[1] = *(const v4u*)(p + 16);
  return f.v;
}

__device__ __forceinline__ v8f wmma_bf16(v16bf a, v16bf b, v8f c) {
  return __builtin_amdgcn_wmma_f32_16x16x32_bf16(
      /*neg_a=*/false, a, /*neg_b=*/false, b,
      /*c_mod=*/(short)0, c, /*reuse_a=*/false, /*reuse_b=*/false);
}

__device__ __forceinline__ float sigmoidf_(float x) {
  return 1.0f / (1.0f + __expf(-x));
}

__device__ __forceinline__ v8f zero8() {
  v8f z = {0.f, 0.f, 0.f, 0.f, 0.f, 0.f, 0.f, 0.f};
  return z;
}

// Cooperative staging: 256 threads copy a 64-row x 128-K bf16 chunk of Bt.
// Each thread owns 32 elements (64B): ldsRow = tid>>2, quarter q = tid&3.
struct StageRegs { v4u r[4]; };

__device__ __forceinline__ void stage_gload(const __bf16* Bt, int gcol, int k0,
                                            int q, StageRegs& s) {
  const v4u* p = (const v4u*)(Bt + gcol * Kd + k0 + q * 32);
  s.r[0] = p[0]; s.r[1] = p[1]; s.r[2] = p[2]; s.r[3] = p[3];
}

__device__ __forceinline__ void stage_sstore(__bf16* buf, int ldsRow, int q,
                                             const StageRegs& s) {
  v4u* d = (v4u*)(buf + ldsRow * LDSP + q * 32);
  d[0] = s.r[0]; d[1] = s.r[1]; d[2] = s.r[2]; d[3] = s.r[3];
}

// ---------------------------------------------------------------------------
// Prologue kernels: convert fp32 weights to bf16, TRANSPOSED to [N x K].
// ---------------------------------------------------------------------------
__global__ void conv_transpose_bf16(const float* __restrict__ src,
                                    __bf16* __restrict__ dst, int K, int N) {
  const int idx = blockIdx.x * blockDim.x + threadIdx.x;   // idx = n*K + k
  if (idx >= K * N) return;
  const int k = idx % K;
  const int n = idx / K;
  dst[idx] = (__bf16)src[k * N + n];
}

// Wct [1024 x 512] = transpose of [Wk ; Wr] (K: 0..255 = x dims, 256..511 = h).
__global__ void build_wct(const float* __restrict__ Wk,
                          const float* __restrict__ Wr,
                          __bf16* __restrict__ dst) {
  const int idx = blockIdx.x * blockDim.x + threadIdx.x;   // idx = n*512 + k
  if (idx >= 1024 * 512) return;
  const int k = idx & 511;
  const int n = idx >> 9;
  const float v = (k < 256) ? Wk[k * 1024 + n] : Wr[(k - 256) * 1024 + n];
  dst[idx] = (__bf16)v;
}

// Gather neighbor states, build cs = [h | gathered] (bf16) and x half of xh.
__global__ void gather_build(const float* __restrict__ x,
                             const float* __restrict__ h,
                             const int* __restrict__ vid,
                             __bf16* __restrict__ cs,
                             __bf16* __restrict__ xh) {
  const int idx = blockIdx.x * blockDim.x + threadIdx.x;
  if (idx >= Bn * Hn) return;
  const int b = idx >> 8;
  const int j = idx & 255;
  cs[b * 512 + j] = (__bf16)h[idx];
  int pi = vid[b] + 1;                  // padded row index (jnp.take clips)
  pi = pi < 0 ? 0 : (pi > Bn ? Bn : pi);
  const float g = (pi == 0) ? 0.0f : h[(pi - 1) * Hn + j];
  cs[b * 512 + Hn + j] = (__bf16)g;
  xh[b * 512 + j] = (__bf16)x[idx];
}

// ---------------------------------------------------------------------------
// Unified GEMM core. Block = 256 threads = 8 waves; wave = 32 rows x 64 cols
// (2 A frags x 4 B tiles -> 8 WMMAs per K-step). B staged via LDS (double
// buffered); next chunk's global loads overlap current chunk's WMMAs.
// MODE 0: C = leaky(A@Bt^T + bias) -> bf16 (cols = blockIdx.y*64 ..)
// MODE 1: layer3 + residual (N=256): fh = vid>=0 ? h + leaky(..) : h
// MODE 2: LSTM: 4 staged 16-col groups are the i/f/g/o slices at
//         col offsets {0,256,512,768}; cell update fused in epilogue.
// ---------------------------------------------------------------------------
template <int MODE>
__global__ __launch_bounds__(256) void gemm_core(
    const __bf16* __restrict__ A, const __bf16* __restrict__ Bt,
    const float* __restrict__ bias, int N,
    __bf16* __restrict__ Cbf,
    const float* __restrict__ h, const int* __restrict__ vid,
    float* __restrict__ fh32, __bf16* __restrict__ xh,
    const float* __restrict__ c_in, const unsigned char* __restrict__ rs_p,
    float* __restrict__ out) {
  __shared__ __bf16 lds[2][64 * LDSP];

  const int lane    = threadIdx.x & 31;
  const int wave    = threadIdx.x >> 5;
  const int rowBase = (blockIdx.x * 8 + wave) * 32;

  // staging role of this thread
  const int ldsRow = threadIdx.x >> 2;          // 0..63
  const int q      = threadIdx.x & 3;           // 32-element quarter
  int gcol;
  if (MODE == 2)
    gcol = blockIdx.y * 16 + ((ldsRow >> 4) << 8) + (ldsRow & 15);
  else
    gcol = blockIdx.y * 64 + ldsRow;

  v8f acc[4][2];
#pragma unroll
  for (int t = 0; t < 4; ++t) { acc[t][0] = zero8(); acc[t][1] = zero8(); }

  constexpr int NCHUNK = Kd / KC;               // 4
  StageRegs sr;
  stage_gload(Bt, gcol, 0, q, sr);

  for (int c = 0; c < NCHUNK; ++c) {
    __bf16* buf = lds[c & 1];
    stage_sstore(buf, ldsRow, q, sr);
    __syncthreads();                            // waits dscnt; publishes chunk
    if (c + 1 < NCHUNK) {
      stage_gload(Bt, gcol, (c + 1) * KC, q, sr);   // overlaps WMMAs below
      __builtin_prefetch(A + (rowBase + (lane & 15)) * Kd + (c + 1) * KC, 0, 1);
    }
#pragma unroll
    for (int kk = 0; kk < KC; kk += 32) {
      const int k0 = c * KC + kk;
      const v16bf a0 = load_a_frag(A, rowBase, k0, lane);
      const v16bf a1 = load_a_frag(A, rowBase + 16, k0, lane);
#pragma unroll
      for (int t = 0; t < 4; ++t) {
        const v16bf b = lds_b_frag(buf, t * 16, kk, lane);
        acc[t][0] = wmma_bf16(a0, b, acc[t][0]);
        acc[t][1] = wmma_bf16(a1, b, acc[t][1]);
      }
    }
    __syncthreads();                            // chunk consumed; buf reusable
  }

  // ---------------- epilogue ----------------
  const int rOff = (lane >> 4) << 3;            // 0 or 8
  const int cl   = lane & 15;

  if (MODE == 0) {
#pragma unroll
    for (int t = 0; t < 4; ++t) {
      const int col = blockIdx.y * 64 + t * 16 + cl;
      const float bv = bias[col];
#pragma unroll
      for (int half = 0; half < 2; ++half) {
#pragma unroll
        for (int j = 0; j < 8; ++j) {
          const int row = rowBase + half * 16 + rOff + j;
          float z = acc[t][half][j] + bv;
          z = z > 0.0f ? z : kLeaky * z;
          Cbf[row * N + col] = (__bf16)z;
        }
      }
    }
  } else if (MODE == 1) {
#pragma unroll
    for (int t = 0; t < 4; ++t) {
      const int col = blockIdx.y * 64 + t * 16 + cl;
      const float bv = bias[col];
#pragma unroll
      for (int half = 0; half < 2; ++half) {
#pragma unroll
        for (int j = 0; j < 8; ++j) {
          const int row = rowBase + half * 16 + rOff + j;
          float z = acc[t][half][j] + bv;
          z = z > 0.0f ? z : kLeaky * z;
          const float hv = h[row * Hn + col];
          const float fh = (vid[row] >= 0) ? hv + z : hv;
          fh32[row * Hn + col] = fh;
          xh[row * Kd + Hn + col] = (__bf16)fh;
        }
      }
    }
  } else {  // MODE 2: LSTM cell update; gates i,f,g,o = tiles 0..3
    const bool rs = rs_p[0] != 0;
    const int col = blockIdx.y * 16 + cl;
    const float bI = bias[col],       bF = bias[col + 256];
    const float bG = bias[col + 512], bO = bias[col + 768];
#pragma unroll
    for (int half = 0; half < 2; ++half) {
#pragma unroll
      for (int j = 0; j < 8; ++j) {
        const int row = rowBase + half * 16 + rOff + j;
        const int off = row * Hn + col;
        const float iv = sigmoidf_(acc[0][half][j] + bI);
        const float fv = sigmoidf_(acc[1][half][j] + bF);
        const float gv = tanhf(acc[2][half][j] + bG);
        const float ov = sigmoidf_(acc[3][half][j] + bO);
        const float co = c_in[off];
        const float cn = fv * co + iv * gv;
        const float hn = ov * tanhf(cn);
        out[off] = hn;                                  // outputs
        out[Bn * Hn + off]     = rs ? hn : fh32[off];   // h_out
        out[2 * Bn * Hn + off] = rs ? cn : co;          // c_out
      }
    }
  }
}

// ---------------------------------------------------------------------------
extern "C" void kernel_launch(void* const* d_in, const int* in_sizes, int n_in,
                              void* d_out, int out_size, void* d_ws, size_t ws_size,
                              hipStream_t stream) {
  const float* x   = (const float*)d_in[0];
  const int*   vid = (const int*)d_in[1];
  const unsigned char* rs = (const unsigned char*)d_in[2];
  const float* h   = (const float*)d_in[3];
  const float* c   = (const float*)d_in[4];
  const float* W1  = (const float*)d_in[5];
  const float* b1  = (const float*)d_in[6];
  const float* W2  = (const float*)d_in[7];
  const float* b2  = (const float*)d_in[8];
  const float* W3  = (const float*)d_in[9];
  const float* b3  = (const float*)d_in[10];
  const float* Wk  = (const float*)d_in[11];
  const float* Wr  = (const float*)d_in[12];
  const float* bl  = (const float*)d_in[13];

  // Workspace layout (~136 MB): transposed bf16 weights, then activation bufs.
  __bf16* w1t  = (__bf16*)d_ws;                       // 512*512
  __bf16* w2t  = w1t + 512 * 512;                     // 512*512
  __bf16* w3t  = w2t + 512 * 512;                     // 256*512
  __bf16* wct  = w3t + 256 * 512;                     // 1024*512
  __bf16* bufA = wct + 1024 * 512;                    // B*512 (cs, then t2)
  __bf16* bufB = bufA + (size_t)Bn * 512;             // B*512 (t1)
  __bf16* xh   = bufB + (size_t)Bn * 512;             // B*512 ([x | final_h])
  float*  fh32 = (float*)(xh + (size_t)Bn * 512);     // B*256 fp32 final_h

  // Prologue: bf16-transpose weights (tiny; L2-resident afterwards).
  conv_transpose_bf16<<<(512 * 512 + 255) / 256, 256, 0, stream>>>(W1, w1t, 512, 512);
  conv_transpose_bf16<<<(512 * 512 + 255) / 256, 256, 0, stream>>>(W2, w2t, 512, 512);
  conv_transpose_bf16<<<(256 * 512 + 255) / 256, 256, 0, stream>>>(W3, w3t, 512, 256);
  build_wct<<<(1024 * 512 + 255) / 256, 256, 0, stream>>>(Wk, Wr, wct);

  // Gather + pack activations.
  gather_build<<<(Bn * Hn + 255) / 256, 256, 0, stream>>>(x, h, vid, bufA, xh);

  // MLP layers (WMMA bf16, f32 accumulate), LDS-staged weights.
  dim3 g1(Bn / 256, 512 / 64);
  gemm_core<0><<<g1, 256, 0, stream>>>(bufA, w1t, b1, 512, bufB,
                                       nullptr, nullptr, nullptr, nullptr,
                                       nullptr, nullptr, nullptr);
  gemm_core<0><<<g1, 256, 0, stream>>>(bufB, w2t, b2, 512, bufA,
                                       nullptr, nullptr, nullptr, nullptr,
                                       nullptr, nullptr, nullptr);
  dim3 g3(Bn / 256, 256 / 64);
  gemm_core<1><<<g3, 256, 0, stream>>>(bufA, w3t, b3, 256, nullptr,
                                       h, vid, fh32, xh,
                                       nullptr, nullptr, nullptr);

  // Gate-fused LSTM GEMM + elementwise cell update.
  dim3 g4(Bn / 256, 256 / 16);
  gemm_core<2><<<g4, 256, 0, stream>>>(xh, wct, bl, 256, nullptr,
                                       nullptr, nullptr, fh32, nullptr,
                                       c, rs, (float*)d_out);
}
